// SelfAttention_31095563223442
// MI455X (gfx1250) — compile-verified, matrix operand order checked by reading, not run on hardware
//
#include <hip/hip_runtime.h>

// ---------------------------------------------------------------------------
// Causal multi-head self-attention fwd for MI455X (gfx1250, wave32, WMMA).
// fp32 in -> bf16 WMMA (f32 accum) -> fp32 out.  Compute-bound: ~43 GFLOP vs
// <100MB of HBM traffic (fits in 192MB L2), so everything funnels through
// v_wmma_f32_16x16x32_bf16, with global_load_async_to_lds_b128 double
// buffering feeding the GEMM.
// ---------------------------------------------------------------------------

#define D_IN      1024
#define D_OUT     1024
#define NUM_HEADS 16
#define HEAD_DIM  64
#define SEQ       2048
#define BATCH     2
#define M_TOTAL   (BATCH * SEQ)   // 4096 rows for the QKV GEMM

typedef __attribute__((ext_vector_type(16))) __bf16       v16bf;
typedef __attribute__((ext_vector_type(8)))  float        v8f;
typedef __attribute__((ext_vector_type(4)))  unsigned int v4u;

union FragU { v4u u[2]; v16bf f; };

__device__ __forceinline__ unsigned short f2bf(float x) {
  unsigned int u = __float_as_uint(x);
  u += 0x7FFFu + ((u >> 16) & 1u);        // round-to-nearest-even
  return (unsigned short)(u >> 16);
}

// Load one 16x32 bf16 A-style fragment (or B-fragment of a K-contiguous
// column-major source).  p = base of this lane's 32-element K run.
// Lane layout per ISA: lanes 0-15 take K {0..7, 16..23}, lanes 16-31 take
// K {8..15, 24..31}, 2 elements per VGPR.
__device__ __forceinline__ v16bf frag_ld(const unsigned short* p, int lane) {
  const int o = (lane >> 4) * 8;
  FragU fu;
  fu.u[0] = *(const v4u*)(p + o);
  fu.u[1] = *(const v4u*)(p + o + 16);
  return fu.f;
}

__device__ __forceinline__ v8f wmma_bf16(v16bf a, v16bf b, v8f c) {
  return __builtin_amdgcn_wmma_f32_16x16x32_bf16(
      /*neg_a=*/false, a, /*neg_b=*/false, b,
      /*c_mod=*/(short)0, c, /*reuse_a=*/false, /*reuse_b=*/false);
}

// Async global->LDS copy, 16B per lane, GV addressing, ASYNCcnt-tracked.
// lds_off = wave-relative LDS byte offset (low 32 bits of generic pointer,
// per ISA 10.2).
__device__ __forceinline__ void async_ld16(unsigned lds_off,
                                           const unsigned short* g) {
  asm volatile("global_load_async_to_lds_b128 %0, %1, off"
               :: "v"(lds_off), "v"(g) : "memory");
}

// ---------------------------------------------------------------------------
// Kernel 1: x (f32, [M_TOTAL, D_IN]) -> bf16, same layout.
// ---------------------------------------------------------------------------
__global__ __launch_bounds__(256) void cvt_x_kernel(const float* __restrict__ x,
                                                    unsigned short* __restrict__ xbf) {
  const int i = blockIdx.x * 256 + threadIdx.x;
  xbf[i] = f2bf(x[i]);
}

// ---------------------------------------------------------------------------
// Kernel 2: W[z] (f32, [K=D_IN, N=D_OUT]) -> bf16 transposed Wt[z][N][K].
// Transposed so the GEMM's B fragments are K-contiguous like A fragments.
// ---------------------------------------------------------------------------
__global__ __launch_bounds__(256) void cvt_w_kernel(const float* __restrict__ Wq,
                                                    const float* __restrict__ Wk,
                                                    const float* __restrict__ Wv,
                                                    unsigned short* __restrict__ wt) {
  const int z = blockIdx.y;
  const int i = blockIdx.x * 256 + threadIdx.x;      // 0 .. 1M-1  (= n*1024 + k)
  const float* W = (z == 0) ? Wq : (z == 1) ? Wk : Wv;
  const int n = i >> 10;
  const int k = i & 1023;
  wt[((size_t)z << 20) + (size_t)i] = f2bf(W[(size_t)k * D_OUT + n]);
}

// ---------------------------------------------------------------------------
// Kernel 3: fused QKV projection GEMM + bias, bf16 WMMA, f32 accum.
//   grid = (N/128, M/128, 3),  block = 256 threads = 8 waves.
//   Workgroup tile 128x128; wave (4x2 grid) owns 32x64 = 8 WMMA accum tiles.
//   Tiles staged with double-buffered global_load_async_to_lds_b128:
//   iteration i consumes buf[i&1] while the 4 per-wave async ops for tile
//   i+1 fill buf[(i+1)&1]; s_wait_asynccnt 0x4 drains only the current tile.
//   Q,K stored [B,H,S,Dh] bf16; V stored transposed [B,H,Dh,S] bf16.
// ---------------------------------------------------------------------------
#define TS_K 32
#define LDSK 40   // padded K stride (shorts); 80B keeps 16B alignment

__global__ __launch_bounds__(256) void qkv_gemm_kernel(
    const unsigned short* __restrict__ xbf,   // [M_TOTAL][D_IN]
    const unsigned short* __restrict__ wt,    // [3][D_OUT][D_IN]
    const float* __restrict__ bq, const float* __restrict__ bk,
    const float* __restrict__ bv,
    unsigned short* __restrict__ qbf,         // [B,H,S,Dh]
    unsigned short* __restrict__ kbf,         // [B,H,S,Dh]
    unsigned short* __restrict__ vbf) {       // [B,H,Dh,S]
  __shared__ unsigned short sA[2][128 * LDSK];
  __shared__ unsigned short sB[2][128 * LDSK];

  const int z     = blockIdx.z;
  const int nBase = blockIdx.x * 128;
  const int mBase = blockIdx.y * 128;
  const int tid   = threadIdx.x;
  const int wid   = tid >> 5;
  const int lane  = tid & 31;
  const int waveM = wid & 3;    // 4 waves along M  (32 rows each)
  const int waveN = wid >> 2;   // 2 waves along N  (64 cols each)
  const int half  = lane >> 4;
  const int nCol  = lane & 15;

  const unsigned short* wz   = wt + ((size_t)z << 20);
  const float*          bias = (z == 0) ? bq : (z == 1) ? bk : bv;

  v8f acc[2][4] = {};

  const int ldRow = tid >> 1;          // 0..127
  const int ldCol = (tid & 1) * 16;    // 0 or 16

  const unsigned short* aRow = &xbf[(size_t)(mBase + ldRow) * D_IN + ldCol];
  const unsigned short* bRow = &wz [(size_t)(nBase + ldRow) * D_IN + ldCol];
  const unsigned aDst0 = (unsigned)(uintptr_t)&sA[0][ldRow * LDSK + ldCol];
  const unsigned bDst0 = (unsigned)(uintptr_t)&sB[0][ldRow * LDSK + ldCol];
  const unsigned bufB  = (unsigned)(128 * LDSK * sizeof(unsigned short));

  // Prologue: tile 0 -> buffer 0 (4 async ops per wave).
  async_ld16(aDst0,      aRow);
  async_ld16(aDst0 + 16, aRow + 8);
  async_ld16(bDst0,      bRow);
  async_ld16(bDst0 + 16, bRow + 8);

  for (int k0 = 0; k0 < D_IN; k0 += TS_K) {
    const int buf = (k0 >> 5) & 1;
    if (k0 + TS_K < D_IN) {
      const unsigned nb = (unsigned)(buf ^ 1) * bufB;
      async_ld16(aDst0 + nb,      aRow + k0 + TS_K);
      async_ld16(aDst0 + nb + 16, aRow + k0 + TS_K + 8);
      async_ld16(bDst0 + nb,      bRow + k0 + TS_K);
      async_ld16(bDst0 + nb + 16, bRow + k0 + TS_K + 8);
      // Drain current tile only; next tile's 4 ops stay in flight.
      asm volatile("s_wait_asynccnt 0x4" ::: "memory");
    } else {
      asm volatile("s_wait_asynccnt 0x0" ::: "memory");
    }
    __syncthreads();

    v16bf aF[2], bF[4];
#pragma unroll
    for (int i = 0; i < 2; ++i)
      aF[i] = frag_ld(&sA[buf][(waveM * 32 + i * 16 + nCol) * LDSK], lane);
#pragma unroll
    for (int j = 0; j < 4; ++j)
      bF[j] = frag_ld(&sB[buf][(waveN * 64 + j * 16 + nCol) * LDSK], lane);

#pragma unroll
    for (int i = 0; i < 2; ++i)
#pragma unroll
      for (int j = 0; j < 4; ++j)
        acc[i][j] = wmma_bf16(aF[i], bF[j], acc[i][j]);
    __syncthreads();   // all reads of buf done before it is refilled (i+2)
  }

  // Epilogue: + bias, cast to bf16, scatter into head-major layouts.
#pragma unroll
  for (int i = 0; i < 2; ++i) {
#pragma unroll
    for (int j = 0; j < 4; ++j) {
      const int   nG = nBase + waveN * 64 + j * 16 + nCol;
      const float bb = bias[nG];
      const int   h  = nG >> 6;
      const int   d  = nG & 63;
#pragma unroll
      for (int r = 0; r < 8; ++r) {
        const int mG = mBase + waveM * 32 + i * 16 + r + 8 * half;
        const int b  = mG >> 11;          // / SEQ
        const int s  = mG & 2047;         // % SEQ
        const unsigned short o16 = f2bf(acc[i][j][r] + bb);
        if (z == 0)
          qbf[(((size_t)b * NUM_HEADS + h) * SEQ + s) * HEAD_DIM + d] = o16;
        else if (z == 1)
          kbf[(((size_t)b * NUM_HEADS + h) * SEQ + s) * HEAD_DIM + d] = o16;
        else
          vbf[(((size_t)b * NUM_HEADS + h) * HEAD_DIM + d) * SEQ + s] = o16;
      }
    }
  }
}

// ---------------------------------------------------------------------------
// Kernel 4: causal flash attention.
//   grid = (SEQ/256, NUM_HEADS, BATCH), block = 256 (8 waves).
//   Each wave owns 32 query rows; streams key tiles of 32 with online softmax.
//   Mask (-1e9) is applied BEFORE the 1/sqrt(Dh) scale, matching the ref.
// ---------------------------------------------------------------------------
#define PPAD 40   // padded key stride for the per-wave P scratch

__global__ __launch_bounds__(256) void attn_kernel(
    const unsigned short* __restrict__ qbf,   // [B,H,S,Dh]
    const unsigned short* __restrict__ kbf,   // [B,H,S,Dh]
    const unsigned short* __restrict__ vbf,   // [B,H,Dh,S]
    float* __restrict__ out) {                // [B,S,D_OUT]
  __shared__ unsigned short sP[8 * 32 * PPAD];   // per-wave P tile (32x32 bf16)

  const int tid  = threadIdx.x;
  const int wid  = tid >> 5;
  const int lane = tid & 31;
  const int half = lane >> 4;
  const int nCol = lane & 15;
  const int b    = blockIdx.z;
  const int h    = blockIdx.y;
  const int q0   = blockIdx.x * 256 + wid * 32;

  const unsigned short* Q = qbf + ((size_t)b * NUM_HEADS + h) * SEQ * HEAD_DIM;
  const unsigned short* K = kbf + ((size_t)b * NUM_HEADS + h) * SEQ * HEAD_DIM;
  const unsigned short* V = vbf + ((size_t)b * NUM_HEADS + h) * HEAD_DIM * SEQ;
  unsigned short* myP = &sP[wid * 32 * PPAD];

  // Q fragments for this wave's 32 rows: [mblock][dh-chunk of 32]
  v16bf qF[2][2];
#pragma unroll
  for (int i = 0; i < 2; ++i)
#pragma unroll
    for (int kk = 0; kk < 2; ++kk)
      qF[i][kk] = frag_ld(&Q[(size_t)(q0 + i * 16 + nCol) * HEAD_DIM + kk * 32], lane);

  v8f   O[2][4] = {};
  float mSt[2][8], lSt[2][8];
#pragma unroll
  for (int i = 0; i < 2; ++i)
#pragma unroll
    for (int r = 0; r < 8; ++r) { mSt[i][r] = -1e30f; lSt[i][r] = 0.0f; }

  const float inv_scale = 0.125f;             // 1/sqrt(64)
  const float masked    = -1e9f * 0.125f;     // ref adds -1e9 then scales
  const int   ktEnd     = (q0 >> 5) + 1;      // causal: keys 0 .. q0+31

  for (int kt = 0; kt < ktEnd; ++kt) {
    const int key0 = kt * 32;

    // ---- S = Q K^T  (M=32, N=32 keys, K=64) ----
    v16bf kF[2][2];
#pragma unroll
    for (int nb = 0; nb < 2; ++nb)
#pragma unroll
      for (int kk = 0; kk < 2; ++kk)
        kF[nb][kk] = frag_ld(&K[(size_t)(key0 + nb * 16 + nCol) * HEAD_DIM + kk * 32], lane);

    v8f sAcc[2][2] = {};
#pragma unroll
    for (int i = 0; i < 2; ++i)
#pragma unroll
      for (int nb = 0; nb < 2; ++nb)
#pragma unroll
        for (int kk = 0; kk < 2; ++kk)
          sAcc[i][nb] = wmma_bf16(qF[i][kk], kF[nb][kk], sAcc[i][nb]);

    // ---- mask + online softmax; write P (bf16) to per-wave LDS ----
#pragma unroll
    for (int i = 0; i < 2; ++i) {
#pragma unroll
      for (int r = 0; r < 8; ++r) {
        const int row = q0 + i * 16 + r + 8 * half;   // C-layout row for this lane
        float s0 = sAcc[i][0][r] * inv_scale;
        float s1 = sAcc[i][1][r] * inv_scale;
        if (key0 + nCol > row)      s0 = masked;
        if (key0 + 16 + nCol > row) s1 = masked;
        float mx = fmaxf(s0, s1);
#pragma unroll
        for (int d = 1; d < 16; d <<= 1) mx = fmaxf(mx, __shfl_xor(mx, d, 32));
        const float mOld  = mSt[i][r];
        const float mNew  = fmaxf(mOld, mx);
        const float alpha = __expf(mOld - mNew);
        mSt[i][r] = mNew;
        const float p0 = __expf(s0 - mNew);
        const float p1 = __expf(s1 - mNew);
        float rs = p0 + p1;
#pragma unroll
        for (int d = 1; d < 16; d <<= 1) rs += __shfl_xor(rs, d, 32);
        lSt[i][r] = lSt[i][r] * alpha + rs;
#pragma unroll
        for (int j = 0; j < 4; ++j) O[i][j][r] *= alpha;
        const int lrow = (i * 16 + r + 8 * half) * PPAD;
        myP[lrow + nCol]      = f2bf(p0);
        myP[lrow + 16 + nCol] = f2bf(p1);
      }
    }
    // per-wave LDS RAW: drain DS before fragment re-read
    asm volatile("s_wait_dscnt 0x0" ::: "memory");

    // ---- O += P V  (M=32, N=64 dh, K=32 keys) ----
    v16bf pF[2], vF[4];
#pragma unroll
    for (int i = 0; i < 2; ++i)
      pF[i] = frag_ld(&myP[(i * 16 + nCol) * PPAD], lane);
#pragma unroll
    for (int j = 0; j < 4; ++j)
      vF[j] = frag_ld(&V[(size_t)(j * 16 + nCol) * SEQ + key0], lane);
#pragma unroll
    for (int i = 0; i < 2; ++i)
#pragma unroll
      for (int j = 0; j < 4; ++j)
        O[i][j] = wmma_bf16(pF[i], vF[j], O[i][j]);
  }

  // ---- normalize and store fp32 ----
#pragma unroll
  for (int i = 0; i < 2; ++i) {
#pragma unroll
    for (int r = 0; r < 8; ++r) {
      const float rcl = 1.0f / lSt[i][r];
      const int   row = q0 + i * 16 + r + 8 * half;
#pragma unroll
      for (int j = 0; j < 4; ++j) {
        const int col = h * HEAD_DIM + j * 16 + nCol;
        out[((size_t)b * SEQ + row) * D_OUT + col] = O[i][j][r] * rcl;
      }
    }
  }
}

// ---------------------------------------------------------------------------
// Host launcher.  Workspace layout (38 MiB total):
//   [0,8M)   xbf   bf16 [4096][1024]
//   [8,14M)  Wt    bf16 [3][1024][1024] (transposed weights)
//   [14,22M) qbf   bf16 [B,H,S,Dh]
//   [22,30M) kbf   bf16 [B,H,S,Dh]
//   [30,38M) vbf   bf16 [B,H,Dh,S]
// ---------------------------------------------------------------------------
extern "C" void kernel_launch(void* const* d_in, const int* in_sizes, int n_in,
                              void* d_out, int out_size, void* d_ws, size_t ws_size,
                              hipStream_t stream) {
  (void)in_sizes; (void)n_in; (void)out_size; (void)ws_size;
  const float* x  = (const float*)d_in[0];
  const float* Wq = (const float*)d_in[1];
  const float* bq = (const float*)d_in[2];
  const float* Wk = (const float*)d_in[3];
  const float* bk = (const float*)d_in[4];
  const float* Wv = (const float*)d_in[5];
  const float* bv = (const float*)d_in[6];
  float* out = (float*)d_out;

  char* ws = (char*)d_ws;
  unsigned short* xbf = (unsigned short*)(ws);
  unsigned short* wt  = (unsigned short*)(ws + ((size_t)8  << 20));
  unsigned short* qbf = (unsigned short*)(ws + ((size_t)14 << 20));
  unsigned short* kbf = (unsigned short*)(ws + ((size_t)22 << 20));
  unsigned short* vbf = (unsigned short*)(ws + ((size_t)30 << 20));

  cvt_x_kernel<<<dim3((M_TOTAL * D_IN) / 256), 256, 0, stream>>>(x, xbf);
  cvt_w_kernel<<<dim3((D_IN * D_OUT) / 256, 3), 256, 0, stream>>>(Wq, Wk, Wv, wt);
  qkv_gemm_kernel<<<dim3(D_OUT / 128, M_TOTAL / 128, 3), 256, 0, stream>>>(
      xbf, wt, bq, bk, bv, qbf, kbf, vbf);
  attn_kernel<<<dim3(SEQ / 256, NUM_HEADS, BATCH), 256, 0, stream>>>(
      qbf, kbf, vbf, out);
}
